// MultiplexInductiveSmoother_34711925686350
// MI455X (gfx1250) — matrix operand order repaired and checked by reading, not compiled, not run on hardware
//
#include <hip/hip_runtime.h>
#include <math.h>

#define P  256
#define DD 128
#define NN 1024
#define EE 250000

typedef __attribute__((ext_vector_type(16))) __bf16 v16bf;
typedef __attribute__((ext_vector_type(8)))  float  v8f;

// ---------- ordered-uint encoding for float atomicMax ----------
__device__ __forceinline__ unsigned f32_ord(float f) {
    unsigned u = __float_as_uint(f);
    return (u & 0x80000000u) ? ~u : (u | 0x80000000u);
}
__device__ __forceinline__ float ord_f32(unsigned e) {
    unsigned u = (e & 0x80000000u) ? (e & 0x7fffffffu) : ~e;
    return __uint_as_float(u);
}

// =====================================================================
// WMMA bf16 GEMM:  C[M,N] = act(A[M,K] @ W[N,K]^T + bias)
// One 16x16 C tile per wave32.  K compile-time, K % 32 == 0; M,N % 16 == 0.
// A-matrix lane layout (16-bit, 16x32): lanes 0-15 K={0..7,16..23},
//                                       lanes 16-31 K={8..15,24..31}
// B-matrix (32x16): same K split as A (lanes 0-15 K={0..7,16..23}, 16-31 K={8..15,24..31}), n = L&15
// C/D (f32): elem r of lane L -> m = r + 8*(L>>4), n = L&15
// =====================================================================
template<bool PRELU, int K>
__global__ void gemm_bf16_wmma(const float* __restrict__ A,
                               const float* __restrict__ W,
                               const float* __restrict__ bias,
                               const float* __restrict__ prelu_a,
                               float* __restrict__ C,
                               int M, int N)
{
    const int wid  = threadIdx.x >> 5;
    const int lane = threadIdx.x & 31;
    const int half = lane >> 4;
    const int l16  = lane & 15;
    const int ntile = N >> 4;
    const int T = (M >> 4) * ntile;
    const int gw = blockIdx.x * (blockDim.x >> 5) + wid;
    if (gw >= T) return;
    const int tm = (gw / ntile) << 4;
    const int tn = (gw % ntile) << 4;

    v8f acc = {};
    const float* Abase = A + (size_t)(tm + l16) * K;
    const float* Wbase = W + (size_t)(tn + l16) * K;

    #pragma unroll 4
    for (int k0 = 0; k0 < K; k0 += 32) {
        const float* ap = Abase + k0 + half * 8;
        float4 a0 = *(const float4*)(ap);
        float4 a1 = *(const float4*)(ap + 4);
        float4 a2 = *(const float4*)(ap + 16);
        float4 a3 = *(const float4*)(ap + 20);
        const float* bp = Wbase + k0 + half * 8;
        float4 b0 = *(const float4*)(bp);
        float4 b1 = *(const float4*)(bp + 4);
        float4 b2 = *(const float4*)(bp + 16);
        float4 b3 = *(const float4*)(bp + 20);

        v16bf av, bvv;
        av[0]=(__bf16)a0.x;  av[1]=(__bf16)a0.y;  av[2]=(__bf16)a0.z;  av[3]=(__bf16)a0.w;
        av[4]=(__bf16)a1.x;  av[5]=(__bf16)a1.y;  av[6]=(__bf16)a1.z;  av[7]=(__bf16)a1.w;
        av[8]=(__bf16)a2.x;  av[9]=(__bf16)a2.y;  av[10]=(__bf16)a2.z; av[11]=(__bf16)a2.w;
        av[12]=(__bf16)a3.x; av[13]=(__bf16)a3.y; av[14]=(__bf16)a3.z; av[15]=(__bf16)a3.w;
        bvv[0]=(__bf16)b0.x;  bvv[1]=(__bf16)b0.y;  bvv[2]=(__bf16)b0.z;  bvv[3]=(__bf16)b0.w;
        bvv[4]=(__bf16)b1.x;  bvv[5]=(__bf16)b1.y;  bvv[6]=(__bf16)b1.z;  bvv[7]=(__bf16)b1.w;
        bvv[8]=(__bf16)b2.x;  bvv[9]=(__bf16)b2.y;  bvv[10]=(__bf16)b2.z; bvv[11]=(__bf16)b2.w;
        bvv[12]=(__bf16)b3.x; bvv[13]=(__bf16)b3.y; bvv[14]=(__bf16)b3.z; bvv[15]=(__bf16)b3.w;

        acc = __builtin_amdgcn_wmma_f32_16x16x32_bf16(
                  false, av, false, bvv, (short)0, acc, false, false);
    }

    const float pa = PRELU ? *prelu_a : 0.0f;
    #pragma unroll
    for (int r = 0; r < 8; ++r) {
        int m = tm + r + half * 8;
        int n = tn + l16;
        float v = acc[r] + bias[n];
        if (PRELU) v = (v >= 0.0f) ? v : pa * v;
        C[(size_t)m * N + n] = v;
    }
}

// =====================================================================
// prep: zt = refine(target); Qv = Wq zt + bq; qk = Wk^T Qv; bkq = bk.Qv
// =====================================================================
__global__ void prep_kernel(const float* __restrict__ tgt,
                            const float* __restrict__ Wr1, const float* __restrict__ br1,
                            const float* __restrict__ ar1,
                            const float* __restrict__ Wr2, const float* __restrict__ br2,
                            const float* __restrict__ Wq,  const float* __restrict__ bq,
                            const float* __restrict__ Wk,  const float* __restrict__ bk,
                            float* __restrict__ zt, float* __restrict__ Qv,
                            float* __restrict__ qk, float* __restrict__ bkq)
{
    __shared__ float sx[P], sh[P], sz[P], sq[DD];
    const int t = threadIdx.x;
    sx[t] = tgt[t];
    __syncthreads();
    const float a = *ar1;
    float acc = br1[t];
    for (int k = 0; k < P; ++k) acc += Wr1[t * P + k] * sx[k];
    sh[t] = (acc >= 0.0f) ? acc : a * acc;
    __syncthreads();
    acc = br2[t];
    for (int k = 0; k < P; ++k) acc += Wr2[t * P + k] * sh[k];
    sz[t] = acc;  zt[t] = acc;
    __syncthreads();
    if (t < DD) {
        acc = bq[t];
        for (int k = 0; k < P; ++k) acc += Wq[t * P + k] * sz[k];
        sq[t] = acc;  Qv[t] = acc;
    }
    __syncthreads();
    if (t < DD) {
        acc = 0.0f;
        for (int i = 0; i < DD; ++i) acc += Wk[i * DD + t] * sq[i];
        qk[t] = acc;
    }
    if (t == 0) {
        float s = 0.0f;
        for (int i = 0; i < DD; ++i) s += bk[i] * sq[i];
        *bkq = s;
    }
}

// =====================================================================
// init: zero accumulators (must be re-done every launch: atomics below)
// =====================================================================
__global__ void init_kernel(float* __restrict__ den, float* __restrict__ gsc,
                            float* __restrict__ G, unsigned* __restrict__ menc)
{
    const int i = blockIdx.x * blockDim.x + threadIdx.x;
    const int stride = gridDim.x * blockDim.x;
    for (int j = i; j < 2 * NN * DD; j += stride) G[j] = 0.0f;
    if (i < 2 * NN) { den[i] = 0.0f; gsc[i] = 0.0f; menc[i] = f32_ord(-INFINITY); }
}

// =====================================================================
// pass A: s_e = (d_e . qk + bkq)/sqrt(DD);  segment max (ordered atomicMax)
// one edge per wave32; lane = 1 float4 of the 512B drug row (coalesced)
// =====================================================================
__global__ void edge_score_kernel(const int* __restrict__ src, const int* __restrict__ dst,
                                  const float* __restrict__ drug,
                                  const float* __restrict__ qk, const float* __restrict__ bkq,
                                  float* __restrict__ sout, unsigned* __restrict__ menc,
                                  int segbase)
{
    const int lane = threadIdx.x & 31;
    const int gw = (blockIdx.x * blockDim.x + threadIdx.x) >> 5;
    const int nw = (gridDim.x * blockDim.x) >> 5;
    const float4 q = ((const float4*)qk)[lane];
    const float bq = *bkq;
    for (int e = gw; e < EE; e += nw) {
        if (e + nw < EE)  // prefetch next row -> global_prefetch_b8
            __builtin_prefetch(drug + (size_t)dst[e + nw] * DD + lane * 4, 0, 0);
        const float4 v = ((const float4*)(drug + (size_t)dst[e] * DD))[lane];
        float part = v.x * q.x + v.y * q.y + v.z * q.z + v.w * q.w;
        #pragma unroll
        for (int off = 16; off > 0; off >>= 1) part += __shfl_xor(part, off, 32);
        if (lane == 0) {
            float s = (part + bq) * 0.08838834764831845f;  // 1/sqrt(128)
            sout[e] = s;
            atomicMax(&menc[segbase + src[e]], f32_ord(s));
        }
    }
}

// =====================================================================
// pass B: ex = exp(s - m[seg]); den += ex; gsc += ex*(y-6); G[seg] += coef*d
// =====================================================================
__global__ void edge_accum_kernel(const int* __restrict__ src, const int* __restrict__ dst,
                                  const float* __restrict__ y,
                                  const float* __restrict__ drug,
                                  const float* __restrict__ sout,
                                  const unsigned* __restrict__ menc,
                                  float* __restrict__ den, float* __restrict__ gsc,
                                  float* __restrict__ G, int segbase)
{
    const int lane = threadIdx.x & 31;
    const int gw = (blockIdx.x * blockDim.x + threadIdx.x) >> 5;
    const int nw = (gridDim.x * blockDim.x) >> 5;
    for (int e = gw; e < EE; e += nw) {
        if (e + nw < EE)
            __builtin_prefetch(drug + (size_t)dst[e + nw] * DD + lane * 4, 0, 0);
        const int sg = segbase + src[e];
        const float m  = ord_f32(menc[sg]);
        const float ex = __expf(sout[e] - m);
        const float coef = ex * (y[e] - 6.0f);
        const float4 v = ((const float4*)(drug + (size_t)dst[e] * DD))[lane];
        float* g = G + (size_t)sg * DD + lane * 4;
        atomicAdd(g + 0, coef * v.x);
        atomicAdd(g + 1, coef * v.y);
        atomicAdd(g + 2, coef * v.z);
        atomicAdd(g + 3, coef * v.w);
        if (lane == 0) { atomicAdd(&den[sg], ex); atomicAdd(&gsc[sg], coef); }
    }
}

// =====================================================================
// msgs epilogue (fp32, output-critical): msgs = (Wv G + bv*gsc)/den
// =====================================================================
__global__ void msgs_kernel(const float* __restrict__ G, const float* __restrict__ Wv,
                            const float* __restrict__ bv, const float* __restrict__ gsc,
                            const float* __restrict__ den, float* __restrict__ out)
{
    __shared__ float g[DD];
    const int m = blockIdx.x, t = threadIdx.x;
    g[t] = G[(size_t)m * DD + t];
    __syncthreads();
    const float dn = den[m];
    float v = 0.0f;
    if (dn > 0.0f) {
        float acc = bv[t] * gsc[m];
        for (int j = 0; j < DD; ++j) acc += Wv[t * DD + j] * g[j];
        v = acc / fmaxf(dn, 1e-30f);
    }
    out[(size_t)m * DD + t] = v;
}

// =====================================================================
// attn MLP: [zt | zn | layer_emb] -> 128 -> 64 -> 1, LeakyReLU(0.2)
// =====================================================================
__global__ void attn_kernel(const float* __restrict__ Z, const float* __restrict__ zt,
                            const float* __restrict__ lemb,
                            const float* __restrict__ Wa1, const float* __restrict__ ba1,
                            const float* __restrict__ Wa2, const float* __restrict__ ba2,
                            const float* __restrict__ Wa3, const float* __restrict__ ba3,
                            float* __restrict__ attn)
{
    __shared__ float x[2 * P + 16], h1[128], h2[64];
    const int r = blockIdx.x, t = threadIdx.x;
    const int lid = (r >= NN) ? 1 : 0;
    x[t] = zt[t];  x[t + 128] = zt[t + 128];
    x[256 + t] = Z[(size_t)r * P + t];  x[256 + 128 + t] = Z[(size_t)r * P + t + 128];
    if (t < 16) x[512 + t] = lemb[lid * 16 + t];
    __syncthreads();
    float acc = ba1[t];
    for (int c = 0; c < 2 * P + 16; ++c) acc += Wa1[t * (2 * P + 16) + c] * x[c];
    h1[t] = (acc >= 0.0f) ? acc : 0.2f * acc;
    __syncthreads();
    if (t < 64) {
        acc = ba2[t];
        for (int c = 0; c < 128; ++c) acc += Wa2[t * 128 + c] * h1[c];
        h2[t] = (acc >= 0.0f) ? acc : 0.2f * acc;
    }
    __syncthreads();
    if (t == 0) {
        acc = ba3[0];
        for (int c = 0; c < 64; ++c) acc += Wa3[c] * h2[c];
        attn[r] = acc;
    }
}

// =====================================================================
// final: softmax over 2N, v_prior, integration MLP, residual, layernorm
// =====================================================================
__global__ void final_kernel(const float* __restrict__ attn, const float* __restrict__ msgs,
                             const float* __restrict__ zt,
                             const float* __restrict__ Wi1, const float* __restrict__ bi1,
                             const float* __restrict__ ai,
                             const float* __restrict__ Wi2, const float* __restrict__ bi2,
                             const float* __restrict__ lng, const float* __restrict__ lnb,
                             float* __restrict__ zout)
{
    __shared__ float w[2 * NN], red[256], vp[DD], u[P], xv[P];
    const int t = threadIdx.x;

    float mx = -INFINITY;
    for (int i = t; i < 2 * NN; i += 256) mx = fmaxf(mx, attn[i]);
    red[t] = mx; __syncthreads();
    for (int s = 128; s > 0; s >>= 1) { if (t < s) red[t] = fmaxf(red[t], red[t + s]); __syncthreads(); }
    const float amax = red[0]; __syncthreads();

    float sm = 0.0f;
    for (int i = t; i < 2 * NN; i += 256) { float e = __expf(attn[i] - amax); w[i] = e; sm += e; }
    red[t] = sm; __syncthreads();
    for (int s = 128; s > 0; s >>= 1) { if (t < s) red[t] += red[t + s]; __syncthreads(); }
    const float denom = red[0]; __syncthreads();

    if (t < DD) {
        float acc = 0.0f;
        for (int r = 0; r < 2 * NN; ++r) acc += w[r] * msgs[(size_t)r * DD + t];
        vp[t] = acc / denom;
    }
    __syncthreads();

    const float a = *ai;
    float acc = bi1[t];
    for (int c = 0; c < DD; ++c) acc += Wi1[t * DD + c] * vp[c];
    u[t] = (acc >= 0.0f) ? acc : a * acc;
    __syncthreads();

    acc = bi2[t];
    for (int c = 0; c < P; ++c) acc += Wi2[t * P + c] * u[c];
    xv[t] = zt[t] + acc;
    __syncthreads();

    red[t] = xv[t]; __syncthreads();
    for (int s = 128; s > 0; s >>= 1) { if (t < s) red[t] += red[t + s]; __syncthreads(); }
    const float mu = red[0] * (1.0f / P); __syncthreads();
    const float d = xv[t] - mu;
    red[t] = d * d; __syncthreads();
    for (int s = 128; s > 0; s >>= 1) { if (t < s) red[t] += red[t + s]; __syncthreads(); }
    const float var = red[0] * (1.0f / P);
    zout[t] = d * rsqrtf(var + 1e-5f) * lng[t] + lnb[t];
}

// =====================================================================
extern "C" void kernel_launch(void* const* d_in, const int* in_sizes, int n_in,
                              void* d_out, int out_size, void* d_ws, size_t ws_size,
                              hipStream_t stream)
{
    const float* tgt   = (const float*)d_in[0];
    const float* formf = (const float*)d_in[1];
    const float* rolef = (const float*)d_in[2];
    const int*   fsrc  = (const int*)  d_in[4];
    const int*   fdst  = (const int*)  d_in[5];
    const float* fy    = (const float*)d_in[6];
    const int*   rsrc  = (const int*)  d_in[8];
    const int*   rdst  = (const int*)  d_in[9];
    const float* ry    = (const float*)d_in[10];
    const float* drug  = (const float*)d_in[11];
    const float* Wr1 = (const float*)d_in[12]; const float* br1 = (const float*)d_in[13];
    const float* ar1 = (const float*)d_in[14];
    const float* Wr2 = (const float*)d_in[15]; const float* br2 = (const float*)d_in[16];
    const float* Wq  = (const float*)d_in[17]; const float* bq  = (const float*)d_in[18];
    const float* Wk  = (const float*)d_in[19]; const float* bk  = (const float*)d_in[20];
    const float* Wv  = (const float*)d_in[21]; const float* bv  = (const float*)d_in[22];
    const float* lemb= (const float*)d_in[23];
    const float* Wa1 = (const float*)d_in[24]; const float* ba1 = (const float*)d_in[25];
    const float* Wa2 = (const float*)d_in[26]; const float* ba2 = (const float*)d_in[27];
    const float* Wa3 = (const float*)d_in[28]; const float* ba3 = (const float*)d_in[29];
    const float* Wi1 = (const float*)d_in[30]; const float* bi1 = (const float*)d_in[31];
    const float* ai  = (const float*)d_in[32];
    const float* Wi2 = (const float*)d_in[33]; const float* bi2 = (const float*)d_in[34];
    const float* lng = (const float*)d_in[35]; const float* lnb = (const float*)d_in[36];

    float* out = (float*)d_out;
    float* ws  = (float*)d_ws;

    // workspace layout (floats)
    float*    ZT   = ws;            // 256
    float*    QV   = ws + 256;      // 128
    float*    QK   = ws + 384;      // 128
    float*    BKQ  = ws + 512;      // 1
    float*    H    = ws + 1024;                  // 2048*256
    float*    Zb   = H + 2048 * 256;             // 2048*256
    float*    S    = Zb + 2048 * 256;            // EE
    unsigned* MENC = (unsigned*)(S + EE);        // 2048
    float*    DEN  = (float*)(MENC + 2048);      // 2048
    float*    GSC  = DEN + 2048;                 // 2048
    float*    G    = GSC + 2048;                 // 2048*128
    float*    ATT  = G + 2048 * 128;             // 2048

    float* zout = out;          // [256]
    float* msgs = out + 256;    // [2048,128] = form_msgs ++ role_msgs (contiguous)

    prep_kernel<<<1, 256, 0, stream>>>(tgt, Wr1, br1, ar1, Wr2, br2, Wq, bq, Wk, bk,
                                       ZT, QV, QK, BKQ);
    init_kernel<<<512, 256, 0, stream>>>(DEN, GSC, G, MENC);

    // refine (WMMA bf16): layer1 with PReLU, layer2 plain
    gemm_bf16_wmma<true,  P><<<128, 256, 0, stream>>>(formf, Wr1, br1, ar1, H,            1024, P);
    gemm_bf16_wmma<true,  P><<<128, 256, 0, stream>>>(rolef, Wr1, br1, ar1, H + 1024 * P, 1024, P);
    gemm_bf16_wmma<false, P><<<256, 256, 0, stream>>>(H,     Wr2, br2, nullptr, Zb,       2048, P);

    // build_msgs: form (segments 0..1023) then role (1024..2047)
    edge_score_kernel<<<2048, 256, 0, stream>>>(fsrc, fdst, drug, QK, BKQ, S, MENC, 0);
    edge_accum_kernel<<<2048, 256, 0, stream>>>(fsrc, fdst, fy, drug, S, MENC, DEN, GSC, G, 0);
    edge_score_kernel<<<2048, 256, 0, stream>>>(rsrc, rdst, drug, QK, BKQ, S, MENC, NN);
    edge_accum_kernel<<<2048, 256, 0, stream>>>(rsrc, rdst, ry, drug, S, MENC, DEN, GSC, G, NN);

    msgs_kernel<<<2048, 128, 0, stream>>>(G, Wv, bv, GSC, DEN, msgs);

    attn_kernel<<<2048, 128, 0, stream>>>(Zb, ZT, lemb, Wa1, ba1, Wa2, ba2, Wa3, ba3, ATT);

    final_kernel<<<1, 256, 0, stream>>>(ATT, msgs, ZT, Wi1, bi1, ai, Wi2, bi2, lng, lnb, zout);
}
